// DecoderRNN_43662637531381
// MI455X (gfx1250) — matrix operand order repaired and checked by reading, not compiled, not run on hardware
//
#include <hip/hip_runtime.h>
#include <hip/hip_bf16.h>

// ---------------------------------------------------------------------------
// Problem constants (from the reference)
// ---------------------------------------------------------------------------
#define BATCH 64
#define P49   49          // GH*GW
#define FDIM  1024
#define ADIM  512
#define EDIM  512
#define HDIM  512
#define VOC   30000
#define TCAP  20
#define TM    19          // TCAP-1 decode steps
#define XDIM  (EDIM + FDIM)   // 1536
#define GDIM  (3*HDIM)        // 1536

// ---------------------------------------------------------------------------
// WMMA types (probe-confirmed signatures for gfx1250)
// ---------------------------------------------------------------------------
typedef __attribute__((ext_vector_type(16))) __bf16 v16bf;
typedef __attribute__((ext_vector_type(8)))  float  v8f;

union FragBF {
    v16bf        v;
    unsigned int d[8];
};

__device__ __forceinline__ unsigned short f2bf(float f) {
    unsigned int u = __float_as_uint(f);
    unsigned int r = (u + 0x7FFFu + ((u >> 16) & 1u)) >> 16;   // RNE
    return (unsigned short)r;
}
__device__ __forceinline__ float sigmoidf_(float x) { return 1.0f / (1.0f + __expf(-x)); }

// ---------------------------------------------------------------------------
// bf16 WMMA GEMM, MT M-tiles per wave (64x16 C strip when MT=4):
//   C[m,n] = (sum_k A[m,k]*W[n,k]) + bias[n]   (* row_scale[m] if given)
// A: M x K bf16 row-major (lda), W: N x K bf16 row-major (ldb), C row stride crs.
// Per K-step: ONE B fragment load feeds MT back-to-back WMMAs -> 4x less
// weight traffic on the L2-resident Wfc stream and 4 independent acc chains.
// A/B fragments use contiguous k = 16*half + s (same order for both); f32 C per ISA 7.12.2.
// ---------------------------------------------------------------------------
template <int MT>
__global__ void wmma_gemm_bf16(const unsigned short* __restrict__ A, int lda,
                               const unsigned short* __restrict__ W, int ldb,
                               const float* __restrict__ bias,
                               const float* __restrict__ row_scale,
                               float* __restrict__ C, long long crs,
                               int M, int N, int K)
{
    const int wave = threadIdx.x >> 5;
    const int gw   = blockIdx.x * (blockDim.x >> 5) + wave;
    const int tiles_n = N >> 4;
    const int tiles_m = M / (16 * MT);
    if (gw >= tiles_m * tiles_n) return;          // wave-uniform exit (EXEC stays full)

    const int tm = gw / tiles_n;
    const int tn = gw - tm * tiles_n;
    const int row0 = tm * (16 * MT);
    const int col0 = tn << 4;

    const int lane = threadIdx.x & 31;
    const int half = lane >> 4;                   // lane group 0..15 / 16..31
    const int l15  = lane & 15;

    const unsigned short* Arow = A + (long long)(row0 + l15) * lda;           // A row = M index
    const unsigned short* Wrow = W + (long long)(col0 + l15) * ldb + 16*half; // B col = N index

    v8f acc[MT] = {};
    for (int k0 = 0; k0 < K; k0 += 32) {
        // B 32x16 bf16: lanes 0-15 K=0..15, lanes 16-31 K=16..31, pairs 2p,2p+1
        FragBF b;
#pragma unroll
        for (int p = 0; p < 8; ++p)
            b.d[p] = *(const unsigned int*)(Wrow + k0 + 2*p);

        if (k0 + 32 < K) {                        // stream next K tile through L2
            __builtin_prefetch(Arow + k0 + 32, 0, 1);
            __builtin_prefetch(Wrow + k0 + 32, 0, 1);
        }

        // A 16x32 bf16: lanes 0-15 K=0..15, lanes 16-31 K=16..31, pairs 2p,2p+1
        FragBF a[MT];
#pragma unroll
        for (int i = 0; i < MT; ++i) {
            const unsigned short* Ai = Arow + (long long)(16 * i) * lda + k0;
#pragma unroll
            for (int p = 0; p < 4; ++p) {
                a[i].d[p]     = *(const unsigned int*)(Ai + 2*p      + 16*half);
                a[i].d[p + 4] = *(const unsigned int*)(Ai + 8 + 2*p  + 16*half);
            }
        }
#pragma unroll
        for (int i = 0; i < MT; ++i)
            acc[i] = __builtin_amdgcn_wmma_f32_16x16x32_bf16(
                /*neg_a=*/false, a[i].v, /*neg_b=*/false, b.v,
                /*c_mod=*/(short)0, acc[i], /*reuse_a=*/false, /*reuse_b=*/false);
    }

    const int n = col0 + l15;
    const float bs = bias ? bias[n] : 0.0f;
#pragma unroll
    for (int i = 0; i < MT; ++i) {
#pragma unroll
        for (int r = 0; r < 8; ++r) {             // C: VGPR r -> row r + 8*half
            const int m = row0 + 16 * i + r + 8*half;
            float val = acc[i][r] + bs;
            if (row_scale) val *= row_scale[m];
            C[(long long)m * crs + n] = val;
        }
    }
}

// ---------------------------------------------------------------------------
// Sort batch by -lengths (stable), emit sort_ind / decode_lengths / caps outputs.
// ---------------------------------------------------------------------------
__global__ void sort_kernel(const int* __restrict__ lengths,
                            const int* __restrict__ captions,
                            int* __restrict__ sort_ind, int* __restrict__ dec_len,
                            float* __restrict__ o_caps, float* __restrict__ o_declen,
                            float* __restrict__ o_sortind)
{
    __shared__ int sh_len[BATCH];
    __shared__ int sh_ind[BATCH];
    const int i = threadIdx.x;
    sh_len[i] = lengths[i];
    __syncthreads();
    const int li = sh_len[i];
    int rank = 0;
    for (int j = 0; j < BATCH; ++j) {
        const int lj = sh_len[j];
        if (lj > li || (lj == li && j < i)) ++rank;
    }
    sh_ind[rank] = i;
    __syncthreads();
    const int si = sh_ind[i];
    sort_ind[i]  = si;
    const int dl = sh_len[si] - 1;
    dec_len[i]   = dl;
    o_declen[i]  = (float)dl;
    o_sortind[i] = (float)si;
    for (int t = 0; t < TCAP; ++t)
        o_caps[i * TCAP + t] = (float)captions[si * TCAP + t];
}

// ---------------------------------------------------------------------------
// Gather features in sorted order (fp32 copy + bf16 copy).
// ---------------------------------------------------------------------------
__global__ void gather_feats_kernel(const float* __restrict__ features,
                                    const int* __restrict__ sort_ind,
                                    float* __restrict__ f32, unsigned short* __restrict__ bf)
{
    const long long n = (long long)BATCH * P49 * FDIM;
    long long i = (long long)blockIdx.x * blockDim.x + threadIdx.x;
    const long long stride = (long long)gridDim.x * blockDim.x;
    for (; i < n; i += stride) {
        const int b = (int)(i / (P49 * FDIM));
        const long long rem = i - (long long)b * (P49 * FDIM);
        const float v = features[(long long)sort_ind[b] * (P49 * FDIM) + rem];
        f32[i] = v;
        bf[i]  = f2bf(v);
    }
}

__global__ void mean_feats_kernel(const float* __restrict__ feats,
                                  unsigned short* __restrict__ mean_bf)
{
    const int b = blockIdx.x;
    for (int f = threadIdx.x; f < FDIM; f += blockDim.x) {
        float s = 0.0f;
        for (int p = 0; p < P49; ++p)
            s += feats[((long long)b * P49 + p) * FDIM + f];
        mean_bf[b * FDIM + f] = f2bf(s * (1.0f / (float)P49));
    }
}

__global__ void f32_to_bf16_kernel(const float* __restrict__ src,
                                   unsigned short* __restrict__ dst, long long n)
{
    long long i = (long long)blockIdx.x * blockDim.x + threadIdx.x;
    const long long stride = (long long)gridDim.x * blockDim.x;
    for (; i < n; i += stride) dst[i] = f2bf(src[i]);
}

// ---------------------------------------------------------------------------
// Fused attention step: e = relu(att1 + att2) . Wf + bf ; softmax over 49 ;
// awe = alpha . feats ; gate = sigmoid(hWbeta) ; x = [emb_t, gate*awe] (bf16) ;
// masked alpha -> d_out.  One block per batch row.
// ---------------------------------------------------------------------------
__global__ void attention_step_kernel(const float* __restrict__ att1,   // (B,49,A)
                                      const float* __restrict__ att2,   // (B,A)  incl bd
                                      const float* __restrict__ Wf,     // (A)
                                      const float* __restrict__ bF,     // (1)
                                      const float* __restrict__ feats,  // (B,49,F) f32
                                      const float* __restrict__ hWbeta, // (B,F)  incl bbeta
                                      const float* __restrict__ emb,    // (V,E)
                                      const int*   __restrict__ captions,
                                      const int*   __restrict__ sort_ind,
                                      const int*   __restrict__ dec_len,
                                      unsigned short* __restrict__ x,   // (B,XDIM) bf16
                                      float* __restrict__ o_alphas,     // (B,TM,49)
                                      int t)
{
    __shared__ float sh_e[P49];
    const int b    = blockIdx.x;
    const int tid  = threadIdx.x;
    const int lane = tid & 31;
    const int wave = tid >> 5;

    // e[p] : one wave per p (8 waves sweep 49 rows)
    for (int p = wave; p < P49; p += 8) {
        const float* a1 = att1 + ((long long)b * P49 + p) * ADIM;
        const float* a2 = att2 + (long long)b * ADIM;
        float s = 0.0f;
        for (int c = lane; c < ADIM; c += 32) {
            float v = a1[c] + a2[c];
            v = v > 0.0f ? v : 0.0f;
            s += v * Wf[c];
        }
        for (int off = 16; off; off >>= 1) s += __shfl_down(s, off, 32);
        if (lane == 0) sh_e[p] = s + bF[0];
    }
    __syncthreads();

    // softmax over 49 (tiny; serial on thread 0)
    if (tid == 0) {
        float m = sh_e[0];
        for (int p = 1; p < P49; ++p) m = fmaxf(m, sh_e[p]);
        float s = 0.0f;
        for (int p = 0; p < P49; ++p) { float v = __expf(sh_e[p] - m); sh_e[p] = v; s += v; }
        const float inv = 1.0f / s;
        for (int p = 0; p < P49; ++p) sh_e[p] *= inv;
    }
    __syncthreads();

    // awe, gate, x[512:1536]
    for (int f = tid; f < FDIM; f += blockDim.x) {
        float aw = 0.0f;
        for (int p = 0; p < P49; ++p)
            aw += sh_e[p] * feats[((long long)b * P49 + p) * FDIM + f];
        const float g = sigmoidf_(hWbeta[(long long)b * FDIM + f]);
        x[(long long)b * XDIM + EDIM + f] = f2bf(g * aw);
    }
    // x[0:512] = embedding of sorted caption token t
    const int cap = captions[sort_ind[b] * TCAP + t];
    for (int c = tid; c < EDIM; c += blockDim.x)
        x[(long long)b * XDIM + c] = f2bf(emb[(long long)cap * EDIM + c]);

    // masked alpha output
    const bool active = t < dec_len[b];
    for (int p = tid; p < P49; p += blockDim.x)
        o_alphas[((long long)b * TM + t) * P49 + p] = active ? sh_e[p] : 0.0f;
}

// ---------------------------------------------------------------------------
// GRU pointwise update + masking; produces h (f32 state), h_bf16 (state),
// hnew_bf16 (pre-mask, feeds the vocab GEMM) and preds row scale.
// ---------------------------------------------------------------------------
__global__ void gru_kernel(const float* __restrict__ gi,   // (B,3H) incl bih
                           const float* __restrict__ gh,   // (B,3H) incl bhh
                           float* __restrict__ h,
                           unsigned short* __restrict__ h_bf,
                           unsigned short* __restrict__ hnew_bf,
                           float* __restrict__ row_scale,
                           const int* __restrict__ dec_len, int t)
{
    const int b = blockIdx.x;
    const bool active = t < dec_len[b];
    for (int c = threadIdx.x; c < HDIM; c += blockDim.x) {
        const long long gb = (long long)b * GDIM;
        const float ir  = gi[gb + c];
        const float iz  = gi[gb + HDIM + c];
        const float inn = gi[gb + 2*HDIM + c];
        const float hr  = gh[gb + c];
        const float hz  = gh[gb + HDIM + c];
        const float hn  = gh[gb + 2*HDIM + c];
        const float r = sigmoidf_(ir + hr);
        const float z = sigmoidf_(iz + hz);
        const float n = tanhf(inn + r * hn);
        const long long hb = (long long)b * HDIM + c;
        const float hold = h[hb];
        const float hnew = (1.0f - z) * n + z * hold;
        hnew_bf[hb] = f2bf(hnew);
        const float hs = active ? hnew : hold;
        h[hb]    = hs;
        h_bf[hb] = f2bf(hs);
    }
    if (threadIdx.x == 0) row_scale[b] = active ? 1.0f : 0.0f;
}

// ---------------------------------------------------------------------------
// Host side
// ---------------------------------------------------------------------------
static inline void launch_gemm(const unsigned short* A, int lda,
                               const unsigned short* W, int ldb,
                               const float* bias, const float* row_scale,
                               float* C, long long crs,
                               int M, int N, int K, hipStream_t s)
{
    if (M % 64 == 0) {
        const int tiles  = (M / 64) * (N / 16);
        const int blocks = (tiles * 32 + 255) / 256;
        wmma_gemm_bf16<4><<<blocks, 256, 0, s>>>(A, lda, W, ldb, bias, row_scale,
                                                 C, crs, M, N, K);
    } else {
        const int tiles  = (M / 16) * (N / 16);
        const int blocks = (tiles * 32 + 255) / 256;
        wmma_gemm_bf16<1><<<blocks, 256, 0, s>>>(A, lda, W, ldb, bias, row_scale,
                                                 C, crs, M, N, K);
    }
}

extern "C" void kernel_launch(void* const* d_in, const int* in_sizes, int n_in,
                              void* d_out, int out_size, void* d_ws, size_t ws_size,
                              hipStream_t stream)
{
    // ---- inputs (setup_inputs order) ----
    const float* features = (const float*)d_in[0];
    const int*   captions = (const int*)  d_in[1];
    const int*   lengths  = (const int*)  d_in[2];
    const float* We    = (const float*)d_in[3];
    const float* be    = (const float*)d_in[4];
    const float* Wd    = (const float*)d_in[5];
    const float* bd    = (const float*)d_in[6];
    const float* Wf    = (const float*)d_in[7];
    const float* bF    = (const float*)d_in[8];
    const float* emb   = (const float*)d_in[9];
    const float* Wih   = (const float*)d_in[10];
    const float* Whh   = (const float*)d_in[11];
    const float* bih   = (const float*)d_in[12];
    const float* bhh   = (const float*)d_in[13];
    const float* Winit = (const float*)d_in[14];
    const float* binit = (const float*)d_in[15];
    const float* Wbeta = (const float*)d_in[16];
    const float* bbeta = (const float*)d_in[17];
    const float* Wfc   = (const float*)d_in[18];
    const float* bfc   = (const float*)d_in[19];
    (void)in_sizes; (void)n_in; (void)out_size; (void)ws_size;

    // ---- outputs (flat, return order) ----
    float* out       = (float*)d_out;
    float* o_preds   = out;                                  // (B,TM,V)
    float* o_caps    = o_preds + (long long)BATCH * TM * VOC;
    float* o_declen  = o_caps + BATCH * TCAP;
    float* o_alphas  = o_declen + BATCH;                     // (B,TM,49)
    float* o_sortind = o_alphas + (long long)BATCH * TM * P49;

    // ---- workspace bump allocator (~67 MB total) ----
    char* wp = (char*)d_ws;
    auto alloc = [&](size_t bytes) -> void* {
        void* r = wp; wp += (bytes + 255) & ~(size_t)255; return r;
    };
    int*            sort_ind = (int*)alloc(BATCH * sizeof(int));
    int*            dec_len  = (int*)alloc(BATCH * sizeof(int));
    float*          feats32  = (float*)alloc((size_t)BATCH * P49 * FDIM * 4);
    unsigned short* featsbf  = (unsigned short*)alloc((size_t)BATCH * P49 * FDIM * 2);
    unsigned short* meanbf   = (unsigned short*)alloc((size_t)BATCH * FDIM * 2);
    float*          att1     = (float*)alloc((size_t)BATCH * P49 * ADIM * 4);
    unsigned short* Webf     = (unsigned short*)alloc((size_t)ADIM * FDIM * 2);
    unsigned short* Wdbf     = (unsigned short*)alloc((size_t)ADIM * HDIM * 2);
    unsigned short* Winitbf  = (unsigned short*)alloc((size_t)HDIM * FDIM * 2);
    unsigned short* Wbetabf  = (unsigned short*)alloc((size_t)FDIM * HDIM * 2);
    unsigned short* Wihbf    = (unsigned short*)alloc((size_t)GDIM * XDIM * 2);
    unsigned short* Whhbf    = (unsigned short*)alloc((size_t)GDIM * HDIM * 2);
    unsigned short* Wfcbf    = (unsigned short*)alloc((size_t)VOC * HDIM * 2);
    float*          h        = (float*)alloc((size_t)BATCH * HDIM * 4);
    unsigned short* hbf      = (unsigned short*)alloc((size_t)BATCH * HDIM * 2);
    unsigned short* hnewbf   = (unsigned short*)alloc((size_t)BATCH * HDIM * 2);
    float*          att2     = (float*)alloc((size_t)BATCH * ADIM * 4);
    float*          hWbeta   = (float*)alloc((size_t)BATCH * FDIM * 4);
    unsigned short* xbf      = (unsigned short*)alloc((size_t)BATCH * XDIM * 2);
    float*          gi       = (float*)alloc((size_t)BATCH * GDIM * 4);
    float*          gh       = (float*)alloc((size_t)BATCH * GDIM * 4);
    float*          rscale   = (float*)alloc(BATCH * sizeof(float));

    // ---- setup phase ----
    sort_kernel<<<1, BATCH, 0, stream>>>(lengths, captions, sort_ind, dec_len,
                                         o_caps, o_declen, o_sortind);
    gather_feats_kernel<<<4096, 256, 0, stream>>>(features, sort_ind, feats32, featsbf);
    mean_feats_kernel<<<BATCH, 256, 0, stream>>>(feats32, meanbf);

    f32_to_bf16_kernel<<<1024, 256, 0, stream>>>(We,    Webf,    (long long)ADIM * FDIM);
    f32_to_bf16_kernel<<<1024, 256, 0, stream>>>(Wd,    Wdbf,    (long long)ADIM * HDIM);
    f32_to_bf16_kernel<<<1024, 256, 0, stream>>>(Winit, Winitbf, (long long)HDIM * FDIM);
    f32_to_bf16_kernel<<<1024, 256, 0, stream>>>(Wbeta, Wbetabf, (long long)FDIM * HDIM);
    f32_to_bf16_kernel<<<1024, 256, 0, stream>>>(Wih,   Wihbf,   (long long)GDIM * XDIM);
    f32_to_bf16_kernel<<<1024, 256, 0, stream>>>(Whh,   Whhbf,   (long long)GDIM * HDIM);
    f32_to_bf16_kernel<<<4096, 256, 0, stream>>>(Wfc,   Wfcbf,   (long long)VOC * HDIM);

    // h0 = mean(feats) @ Winit^T + binit
    launch_gemm(meanbf, FDIM, Winitbf, FDIM, binit, nullptr, h, HDIM,
                BATCH, HDIM, FDIM, stream);
    f32_to_bf16_kernel<<<64, 256, 0, stream>>>(h, hbf, (long long)BATCH * HDIM);

    // att1 = feats @ We^T + be  (3136 x 512 x 1024)
    launch_gemm(featsbf, FDIM, Webf, FDIM, be, nullptr, att1, ADIM,
                BATCH * P49, ADIM, FDIM, stream);

    // ---- decode scan ----
    for (int t = 0; t < TM; ++t) {
        // att2 = h @ Wd^T + bd
        launch_gemm(hbf, HDIM, Wdbf, HDIM, bd, nullptr, att2, ADIM,
                    BATCH, ADIM, HDIM, stream);
        // hWbeta = h @ Wbeta^T + bbeta (pre-sigmoid)
        launch_gemm(hbf, HDIM, Wbetabf, HDIM, bbeta, nullptr, hWbeta, FDIM,
                    BATCH, FDIM, HDIM, stream);
        // fused attention / gate / x-assembly / alpha output
        attention_step_kernel<<<BATCH, 256, 0, stream>>>(
            att1, att2, Wf, bF, feats32, hWbeta, emb, captions, sort_ind,
            dec_len, xbf, o_alphas, t);
        // gi = x @ Wih^T + bih
        launch_gemm(xbf, XDIM, Wihbf, XDIM, bih, nullptr, gi, GDIM,
                    BATCH, GDIM, XDIM, stream);
        // gh = h @ Whh^T + bhh
        launch_gemm(hbf, HDIM, Whhbf, HDIM, bhh, nullptr, gh, GDIM,
                    BATCH, GDIM, HDIM, stream);
        // GRU pointwise + masking
        gru_kernel<<<BATCH, 256, 0, stream>>>(gi, gh, h, hbf, hnewbf,
                                              rscale, dec_len, t);
        // preds = (h_new @ Wfc^T + bfc) * active   -> directly into d_out
        launch_gemm(hnewbf, HDIM, Wfcbf, HDIM, bfc, rscale,
                    o_preds + (long long)t * VOC, (long long)TM * VOC,
                    BATCH, VOC, HDIM, stream);
    }
}
